// LSDE_23175643529382
// MI455X (gfx1250) — matrix operand order repaired and checked by reading, not compiled, not run on hardware
//
#include <hip/hip_runtime.h>
#include <hip/hip_bf16.h>

// ---------------------------------------------------------------------------
// Neural SDE scan for MI455X (gfx1250).
//   B=256, T=200, D_IN=64, H=512, D_OUT=64, N_CLS=10
// Persistent kernel: 16 workgroups x 256 threads (8 waves, wave32).
// Each WG owns 16 batch rows and runs the full 199-step recurrence with
// v_wmma_f32_16x16x32_f16 (fp16 A/B, fp32 accum), state + intermediates in LDS.
// ---------------------------------------------------------------------------

#define B_SZ   256
#define T_SZ   200
#define D_IN   64
#define H_SZ   512
#define D_OUT  64
#define N_CLS  10
#define LIPC   0.909f

typedef _Float16 h8   __attribute__((ext_vector_type(8)));
typedef _Float16 v16h __attribute__((ext_vector_type(16)));
typedef float    v8f  __attribute__((ext_vector_type(8)));
typedef float    f4   __attribute__((ext_vector_type(4)));

__device__ __forceinline__ float lipswish_f(float x) {
    return LIPC * x / (1.0f + __expf(-x));
}
__device__ __forceinline__ v8f vzero() {
    v8f z = {0.f,0.f,0.f,0.f,0.f,0.f,0.f,0.f};
    return z;
}
__device__ __forceinline__ v16h make_a(const _Float16* p) {
    // A 16x32 f16 fragment: halves 0..15 = K c..c+15 (contiguous, matches B)
    h8 lo = *(const h8*)(p);
    h8 hi = *(const h8*)(p + 8);
    return __builtin_shufflevector(lo, hi, 0,1,2,3,4,5,6,7,8,9,10,11,12,13,14,15);
}

// ---------------------------------------------------------------------------
// Prep kernels (tiny, VALU): weight transpose+fp16 convert, g-path, y0
// ---------------------------------------------------------------------------
__global__ void transpose_f16_kernel(const float* __restrict__ W,
                                     _Float16* __restrict__ Wt, int K, int N) {
    int i = blockIdx.x * blockDim.x + threadIdx.x;
    if (i < K * N) {
        int n = i / K, k = i % K;
        Wt[i] = (_Float16)W[(size_t)k * N + n];     // Wt[n][k] contiguous in k
    }
}

__global__ void gpath1_kernel(const float* __restrict__ times,
                              const float* __restrict__ W_noise,
                              const float* __restrict__ b_noise,
                              const float* __restrict__ Wg1,
                              const float* __restrict__ bg1,
                              float* __restrict__ hAct) {
    int i = blockIdx.x * blockDim.x + threadIdx.x;
    if (i < (T_SZ - 1) * H_SZ) {
        int t = i >> 9, n = i & (H_SZ - 1);
        float tv = times[t];
        float s = bg1[n];
        for (int k = 0; k < H_SZ; ++k) {
            float ttk = tv * W_noise[k] + b_noise[k];
            s += ttk * Wg1[k * H_SZ + n];
        }
        hAct[i] = lipswish_f(s);
    }
}

__global__ void gpath2_kernel(const float* __restrict__ hAct,
                              const float* __restrict__ Wg2,
                              const float* __restrict__ bg2,
                              float* __restrict__ g) {
    int i = blockIdx.x * blockDim.x + threadIdx.x;
    if (i < (T_SZ - 1) * H_SZ) {
        int t = i >> 9, n = i & (H_SZ - 1);
        float s = bg2[n];
        const float* hr = hAct + (size_t)t * H_SZ;
        for (int k = 0; k < H_SZ; ++k) s += hr[k] * Wg2[k * H_SZ + n];
        g[i] = s;
    }
}

__global__ void y0_kernel(const float* __restrict__ coeffs,
                          const float* __restrict__ W_init,
                          const float* __restrict__ b_init,
                          float* __restrict__ y0) {
    int i = blockIdx.x * blockDim.x + threadIdx.x;
    if (i < B_SZ * H_SZ) {
        int b = i >> 9, n = i & (H_SZ - 1);
        const float* xr = coeffs + (size_t)b * (T_SZ - 1) * (4 * D_IN); // a-part, t=0
        float s = b_init[n];
        for (int k = 0; k < D_IN; ++k) s += xr[k] * W_init[k * H_SZ + n];
        y0[i] = s;
    }
}

// ---------------------------------------------------------------------------
// Persistent scan kernel
// Dynamic LDS layout (bytes):
//   [0,32768)        yf32   : 16x512 f32 master state
//   [32768,65536)    zfin   : 16x512 f32 state at last_idx
//   [65536,98304)    concat : 16x1024 f16  [y | xw]
//   [98304,114688)   zA     : 16x512 f16
//   [114688,131072)  zB     : 16x512 f16
//   [131072,131136)  lastIdx: 16 x int
// ---------------------------------------------------------------------------
#define SMEM_BYTES 131136

__global__ __launch_bounds__(256)
void sde_scan_kernel(const float* __restrict__ coeffs,
                     const float* __restrict__ times,
                     const float* __restrict__ noise,
                     const float* __restrict__ b_X,
                     const float* __restrict__ b_emb,
                     const float* __restrict__ bf1,
                     const float* __restrict__ bf2,
                     const float* __restrict__ b_out,
                     const float* __restrict__ b_dec,
                     const float* __restrict__ W_cls,
                     const float* __restrict__ b_cls,
                     const int*   __restrict__ mask,
                     const _Float16* __restrict__ WXT,    // [512][64]
                     const _Float16* __restrict__ WembT,  // [512][1024]
                     const _Float16* __restrict__ Wf1T,   // [512][512]
                     const _Float16* __restrict__ Wf2T,   // [512][512]
                     const _Float16* __restrict__ WoutT,  // [512][512]
                     const _Float16* __restrict__ WdecT,  // [64][512]
                     const float* __restrict__ gbuf,      // [199][512]
                     const float* __restrict__ y0buf,     // [256][512]
                     float* __restrict__ out_pred,        // [256][200][64]
                     float* __restrict__ out_logits) {    // [256][10]
    extern __shared__ char smem[];
    float*    yf32    = (float*)smem;
    float*    zfin    = (float*)(smem + 32768);
    _Float16* concatb = (_Float16*)(smem + 65536);
    _Float16* zA      = (_Float16*)(smem + 98304);
    _Float16* zB      = (_Float16*)(smem + 114688);
    int*      lastIdx = (int*)(smem + 131072);

    const int tid   = threadIdx.x;
    const int lane  = tid & 31;
    const int wave  = tid >> 5;
    const int wg0   = blockIdx.x * 16;                 // first batch row of this WG
    const int nlan  = lane & 15;
    const int cA    = (lane < 16) ? 0 : 16;            // A fragment K sub-offset
    const int cB    = (lane < 16) ? 0 : 16;            // B fragment K sub-offset
    const int rbase = (lane >> 4) << 3;                // C fragment row base

    // ---- per-row last index from mask ----
    if (tid < 16) {
        int len = 0;
        const int* mrow = mask + (size_t)(wg0 + tid) * T_SZ;
        for (int t = 0; t < T_SZ; ++t) len += mrow[t];
        lastIdx[tid] = len - 1;
    }
    // ---- load y0 into LDS (fp32 master + fp16 A-copy) ----
    for (int idx = tid; idx < 16 * H_SZ; idx += 256) {
        int row = idx >> 9, k = idx & (H_SZ - 1);
        float v = y0buf[(size_t)(wg0 + row) * H_SZ + k];
        yf32[idx] = v;
        concatb[row * 1024 + k] = (_Float16)v;
    }
    __syncthreads();

    // ---- fused decoder: out[:, tstep, :] = y @ W_dec + b_dec (waves 0..3),
    //      plus z_final capture (waves 4..7) ----
    auto decode = [&](int tstep) {
        if (wave < 4) {
            const _Float16* aB = concatb + nlan * 1024 + cA;   // y-part, K=512
            int nc = wave * 16 + nlan;                         // 0..63
            const _Float16* bB = WdecT + (size_t)nc * H_SZ + cB;
            v8f acc = vzero();
            for (int kt = 0; kt < 16; ++kt) {
                v16h a = make_a(aB + kt * 32);
                v16h b = *(const v16h*)(bB + kt * 32);
                acc = __builtin_amdgcn_wmma_f32_16x16x32_f16(
                        false, a, false, b, (short)0, acc, false, false);
            }
            float bd = b_dec[nc];
#pragma unroll
            for (int r = 0; r < 8; ++r) {
                int brow = wg0 + rbase + r;
                out_pred[((size_t)brow * T_SZ + tstep) * D_OUT + nc] = acc[r] + bd;
            }
        } else {
            int lt = tid - 128;
            for (int r = 0; r < 16; ++r)
                if (lastIdx[r] == tstep)
                    for (int k = lt; k < H_SZ; k += 128)
                        zfin[r * H_SZ + k] = yf32[r * H_SZ + k];
        }
    };

    // ---- generic dense stage: out = act(A_lds @ W^T + bias), N=512 ----
    auto dense_stage = [&](const _Float16* aLds, int aStride, int kIters,
                           const _Float16* WT, int wStride, const float* bias,
                           _Float16* outLds, int act) {
        const _Float16* aB = aLds + nlan * aStride + cA;
        const _Float16* bB[4];
        v8f acc[4];
#pragma unroll
        for (int j = 0; j < 4; ++j) {
            int nc = (wave + 8 * j) * 16 + nlan;
            bB[j] = WT + (size_t)nc * wStride + cB;
            acc[j] = vzero();
        }
        for (int kt = 0; kt < kIters; ++kt) {
            v16h a = make_a(aB + kt * 32);
#pragma unroll
            for (int j = 0; j < 4; ++j) {
                v16h b = *(const v16h*)(bB[j] + kt * 32);
                acc[j] = __builtin_amdgcn_wmma_f32_16x16x32_f16(
                            false, a, false, b, (short)0, acc[j], false, false);
            }
        }
#pragma unroll
        for (int j = 0; j < 4; ++j) {
            int nc = (wave + 8 * j) * 16 + nlan;
            float bv = bias[nc];
#pragma unroll
            for (int r = 0; r < 8; ++r) {
                float v = acc[j][r] + bv;
                if (act) v = lipswish_f(v);
                outLds[(rbase + r) * H_SZ + nc] = (_Float16)v;
            }
        }
    };

    // ---- xw stage: concat[:,512:1024] = X_t @ W_X + b_X  (K=64, A from global) ----
    auto xw_stage = [&](int t) {
        const float* arow = coeffs + ((size_t)(wg0 + nlan) * (T_SZ - 1) + t) * (4 * D_IN);
        v16h afr[2];
#pragma unroll
        for (int kt = 0; kt < 2; ++kt) {
            int c = cA + kt * 32;
            f4 a0 = *(const f4*)(arow + c);
            f4 a1 = *(const f4*)(arow + c + 4);
            f4 a2 = *(const f4*)(arow + c + 8);
            f4 a3 = *(const f4*)(arow + c + 12);
            v16h a;
#pragma unroll
            for (int i = 0; i < 4; ++i) {
                a[i]      = (_Float16)a0[i];
                a[4 + i]  = (_Float16)a1[i];
                a[8 + i]  = (_Float16)a2[i];
                a[12 + i] = (_Float16)a3[i];
            }
            afr[kt] = a;
        }
#pragma unroll
        for (int j = 0; j < 4; ++j) {
            int nc = (wave + 8 * j) * 16 + nlan;
            const _Float16* bB = WXT + (size_t)nc * D_IN + cB;
            v8f acc = vzero();
#pragma unroll
            for (int kt = 0; kt < 2; ++kt) {
                v16h b = *(const v16h*)(bB + kt * 32);
                acc = __builtin_amdgcn_wmma_f32_16x16x32_f16(
                        false, afr[kt], false, b, (short)0, acc, false, false);
            }
            float bv = b_X[nc];
#pragma unroll
            for (int r = 0; r < 8; ++r)
                concatb[(rbase + r) * 1024 + H_SZ + nc] = (_Float16)(acc[r] + bv);
        }
    };

    // ---- drift + Euler-Maruyama state update ----
    auto drift_stage = [&](int t) {
        float dt = times[t + 1] - times[t];
        float sq = sqrtf(dt);
        const _Float16* aB = zA + nlan * H_SZ + cA;
        const _Float16* bB[4];
        v8f acc[4];
#pragma unroll
        for (int j = 0; j < 4; ++j) {
            int nc = (wave + 8 * j) * 16 + nlan;
            bB[j] = WoutT + (size_t)nc * H_SZ + cB;
            acc[j] = vzero();
        }
        for (int kt = 0; kt < 16; ++kt) {
            v16h a = make_a(aB + kt * 32);
#pragma unroll
            for (int j = 0; j < 4; ++j) {
                v16h b = *(const v16h*)(bB[j] + kt * 32);
                acc[j] = __builtin_amdgcn_wmma_f32_16x16x32_f16(
                            false, a, false, b, (short)0, acc[j], false, false);
            }
        }
#pragma unroll
        for (int j = 0; j < 4; ++j) {
            int nc = (wave + 8 * j) * 16 + nlan;
            float bo = b_out[nc];
            float gv = gbuf[(size_t)t * H_SZ + nc];
#pragma unroll
            for (int r = 0; r < 8; ++r) {
                int row = rbase + r;
                float dw = gv * noise[((size_t)t * B_SZ + (wg0 + row)) * H_SZ + nc] * sq;
                float yn = yf32[row * H_SZ + nc] + (acc[j][r] + bo) * dt + dw;
                yf32[row * H_SZ + nc] = yn;
                concatb[row * 1024 + nc] = (_Float16)yn;
            }
        }
    };

    // ---- run ----
    decode(0);
    for (int t = 0; t < T_SZ - 1; ++t) {
        __syncthreads();
        xw_stage(t);
        __syncthreads();
        dense_stage(concatb, 1024, 32, WembT, 1024, b_emb, zA, 0); // z1
        __syncthreads();
        dense_stage(zA, 512, 16, Wf1T, 512, bf1, zB, 1);           // lipswish hidden
        __syncthreads();
        dense_stage(zB, 512, 16, Wf2T, 512, bf2, zA, 0);           // z3
        __syncthreads();
        drift_stage(t);
        __syncthreads();
        decode(t + 1);
    }
    __syncthreads();

    // ---- logits = z_final @ W_cls + b_cls ----
    if (tid < 16 * N_CLS) {
        int r = tid / N_CLS, c = tid % N_CLS;
        const float* zf = zfin + r * H_SZ;
        float s = b_cls[c];
        for (int k = 0; k < H_SZ; ++k) s += zf[k] * W_cls[k * N_CLS + c];
        out_logits[(wg0 + r) * N_CLS + c] = s;
    }
}

// ---------------------------------------------------------------------------
extern "C" void kernel_launch(void* const* d_in, const int* in_sizes, int n_in,
                              void* d_out, int out_size, void* d_ws, size_t ws_size,
                              hipStream_t stream) {
    (void)in_sizes; (void)n_in; (void)out_size; (void)ws_size;
    const float* coeffs  = (const float*)d_in[0];
    const float* times   = (const float*)d_in[1];
    const float* noise   = (const float*)d_in[2];
    const float* W_X     = (const float*)d_in[3];
    const float* b_X     = (const float*)d_in[4];
    const float* W_emb   = (const float*)d_in[5];
    const float* b_emb   = (const float*)d_in[6];
    const float* Wf1     = (const float*)d_in[7];
    const float* bf1     = (const float*)d_in[8];
    const float* Wf2     = (const float*)d_in[9];
    const float* bf2     = (const float*)d_in[10];
    const float* W_out   = (const float*)d_in[11];
    const float* b_out   = (const float*)d_in[12];
    const float* W_noise = (const float*)d_in[13];
    const float* b_noise = (const float*)d_in[14];
    const float* Wg1     = (const float*)d_in[15];
    const float* bg1     = (const float*)d_in[16];
    const float* Wg2     = (const float*)d_in[17];
    const float* bg2     = (const float*)d_in[18];
    const float* W_init  = (const float*)d_in[19];
    const float* b_init  = (const float*)d_in[20];
    const float* W_dec   = (const float*)d_in[21];
    const float* b_dec   = (const float*)d_in[22];
    const float* W_cls   = (const float*)d_in[23];
    const float* b_cls   = (const float*)d_in[24];
    const int*   mask    = (const int*)d_in[25];

    // workspace carve (256B aligned slots)
    size_t off = 0;
    char* base = (char*)d_ws;
    auto carve = [&](size_t bytes) {
        void* p = base + off;
        off += (bytes + 255) & ~(size_t)255;
        return p;
    };
    _Float16* WXT   = (_Float16*)carve((size_t)H_SZ * D_IN * 2);
    _Float16* WembT = (_Float16*)carve((size_t)H_SZ * 2 * H_SZ * 2);
    _Float16* Wf1T  = (_Float16*)carve((size_t)H_SZ * H_SZ * 2);
    _Float16* Wf2T  = (_Float16*)carve((size_t)H_SZ * H_SZ * 2);
    _Float16* WoutT = (_Float16*)carve((size_t)H_SZ * H_SZ * 2);
    _Float16* WdecT = (_Float16*)carve((size_t)D_OUT * H_SZ * 2);
    float*    hAct  = (float*)carve((size_t)(T_SZ - 1) * H_SZ * 4);
    float*    gbuf  = (float*)carve((size_t)(T_SZ - 1) * H_SZ * 4);
    float*    y0buf = (float*)carve((size_t)B_SZ * H_SZ * 4);

    const int thr = 256;
    auto grid = [&](int n) { return dim3((n + thr - 1) / thr); };

    // prep: weight transpose + fp16 convert (B operand wants [n][k] contiguous k)
    transpose_f16_kernel<<<grid(D_IN * H_SZ),     thr, 0, stream>>>(W_X,   WXT,   D_IN,     H_SZ);
    transpose_f16_kernel<<<grid(2 * H_SZ * H_SZ), thr, 0, stream>>>(W_emb, WembT, 2 * H_SZ, H_SZ);
    transpose_f16_kernel<<<grid(H_SZ * H_SZ),     thr, 0, stream>>>(Wf1,   Wf1T,  H_SZ,     H_SZ);
    transpose_f16_kernel<<<grid(H_SZ * H_SZ),     thr, 0, stream>>>(Wf2,   Wf2T,  H_SZ,     H_SZ);
    transpose_f16_kernel<<<grid(H_SZ * H_SZ),     thr, 0, stream>>>(W_out, WoutT, H_SZ,     H_SZ);
    transpose_f16_kernel<<<grid(H_SZ * D_OUT),    thr, 0, stream>>>(W_dec, WdecT, H_SZ,     D_OUT);

    // prep: diffusion path g(t) and y0
    gpath1_kernel<<<grid((T_SZ - 1) * H_SZ), thr, 0, stream>>>(times, W_noise, b_noise, Wg1, bg1, hAct);
    gpath2_kernel<<<grid((T_SZ - 1) * H_SZ), thr, 0, stream>>>(hAct, Wg2, bg2, gbuf);
    y0_kernel<<<grid(B_SZ * H_SZ), thr, 0, stream>>>(coeffs, W_init, b_init, y0buf);

    // persistent scan: one WG per 16 batch rows
    float* out_pred   = (float*)d_out;
    float* out_logits = out_pred + (size_t)B_SZ * T_SZ * D_OUT;
    sde_scan_kernel<<<dim3(B_SZ / 16), dim3(256), SMEM_BYTES, stream>>>(
        coeffs, times, noise, b_X, b_emb, bf1, bf2, b_out, b_dec, W_cls, b_cls,
        mask, WXT, WembT, Wf1T, Wf2T, WoutT, WdecT, gbuf, y0buf,
        out_pred, out_logits);
}
